// NonLocalBlock_45681272160284
// MI455X (gfx1250) — matrix operand order repaired and checked by reading, not compiled, not run on hardware
//
#include <hip/hip_runtime.h>

typedef unsigned short u16;
typedef __attribute__((ext_vector_type(16))) __bf16 v16bf;
typedef __attribute__((ext_vector_type(8)))  float  v8f;
typedef __attribute__((ext_vector_type(4)))  unsigned int u32x4;

union FragAB { u32x4 u[2]; v16bf v; unsigned int w[8]; };

#define NPIX   9216
#define CCH    64
#define DHEAD  32
#define BM     64     // query rows per workgroup (4 waves x 16)
#define BN     64     // keys per inner iteration
#define LOG2E  1.4426950408889634f

static __device__ __forceinline__ u16 f32_to_bf16(float f) {
  unsigned int u = __float_as_uint(f);
  u += 0x7FFFu + ((u >> 16) & 1u);
  return (u16)(u >> 16);
}
static __device__ __forceinline__ unsigned int pack_bf16x2(float lo, float hi) {
  return (unsigned int)f32_to_bf16(lo) | ((unsigned int)f32_to_bf16(hi) << 16);
}
// truncating bf16 pair pack: single v_perm_b32
static __device__ __forceinline__ unsigned int pack_bf16x2_trunc(float lo, float hi) {
  return __builtin_amdgcn_perm(__float_as_uint(hi), __float_as_uint(lo), 0x07060302u);
}
// raw v_exp_f32 (2^x, flushes deep-negative to 0 -- exactly right for softmax)
static __device__ __forceinline__ float exp2_raw(float x) {
  return __builtin_amdgcn_exp2f(x);
}
// lane <-> lane^16 exchange with pre-hoisted byte address
static __device__ __forceinline__ float xor16_f(float v, int bp) {
  return __int_as_float(__builtin_amdgcn_ds_bpermute(bp, __float_as_int(v)));
}
static __device__ __forceinline__ unsigned xor16_u(unsigned v, int bp) {
  return (unsigned)__builtin_amdgcn_ds_bpermute(bp, (int)v);
}

// ---------------------------------------------------------------------------
// Kernel 1: fused 1x1-conv projections.
//  Q (phi)   : [B][N][32] bf16, pre-scaled by log2(e) so flash softmax can use
//              raw v_exp_f32 (exp2) with no per-element multiply.
//  K (theta) : [B][N][32] bf16  (token-major -> WMMA A-fragment friendly)
//  V (g)     : [B][32][N] bf16  (i-major -> PV A-fragment friendly)
// ---------------------------------------------------------------------------
__global__ __launch_bounds__(256)
void proj_kernel(const float* __restrict__ x,
                 const float* __restrict__ g_w, const float* __restrict__ g_b,
                 const float* __restrict__ th_w, const float* __restrict__ th_b,
                 const float* __restrict__ ph_w, const float* __restrict__ ph_b,
                 u16* __restrict__ Q, u16* __restrict__ K, u16* __restrict__ V)
{
  __shared__ float xs[CCH * 256];
  __shared__ float ws[3 * DHEAD * CCH];   // theta | phi | g
  __shared__ float bs[3 * DHEAD];

  const int b   = blockIdx.y;
  const int p0  = blockIdx.x * 256;
  const int tid = threadIdx.x;

  for (int i = tid; i < CCH * 256; i += 256)
    xs[i] = x[((size_t)(b * CCH + (i >> 8))) * NPIX + p0 + (i & 255)];
  for (int i = tid; i < DHEAD * CCH; i += 256) {
    ws[i]               = th_w[i];
    ws[DHEAD*CCH + i]   = ph_w[i];
    ws[2*DHEAD*CCH + i] = g_w[i];
  }
  if (tid < DHEAD) {
    bs[tid]           = th_b[tid];
    bs[DHEAD + tid]   = ph_b[tid];
    bs[2*DHEAD + tid] = g_b[tid];
  }
  __syncthreads();

  float xr[CCH];
  #pragma unroll
  for (int c = 0; c < CCH; ++c) xr[c] = xs[c * 256 + tid];

  float thv[DHEAD], phv[DHEAD], gv[DHEAD];
  for (int o = 0; o < DHEAD; ++o) {
    float at = bs[o], ap = bs[DHEAD + o], ag = bs[2*DHEAD + o];
    #pragma unroll
    for (int c = 0; c < CCH; ++c) {
      const float xv = xr[c];
      at += ws[o*CCH + c] * xv;
      ap += ws[DHEAD*CCH + o*CCH + c] * xv;
      ag += ws[2*DHEAD*CCH + o*CCH + c] * xv;
    }
    thv[o] = at; phv[o] = ap * LOG2E; gv[o] = ag;
  }

  const size_t p = (size_t)(b * NPIX) + p0 + tid;
  unsigned int* Qp = (unsigned int*)(Q + p * DHEAD);
  unsigned int* Kp = (unsigned int*)(K + p * DHEAD);
  #pragma unroll
  for (int o2 = 0; o2 < DHEAD/2; ++o2) {
    Qp[o2] = pack_bf16x2(phv[2*o2], phv[2*o2+1]);
    Kp[o2] = pack_bf16x2(thv[2*o2], thv[2*o2+1]);
  }
  #pragma unroll
  for (int o = 0; o < DHEAD; ++o)
    V[((size_t)(b * DHEAD + o)) * NPIX + p0 + tid] = f32_to_bf16(gv[o]);
}

// ---------------------------------------------------------------------------
// Kernel 2: flash-attention (S^T formulation, log2-domain softmax) + fused
// output conv + residual. Keys streamed via async-to-LDS double buffering.
// ---------------------------------------------------------------------------
__global__ __launch_bounds__(128)
void flash_kernel(const u16* __restrict__ Qg, const u16* __restrict__ Kg,
                  const u16* __restrict__ Vg, const float* __restrict__ x,
                  const float* __restrict__ Ww, const float* __restrict__ Wb,
                  float* __restrict__ out)
{
  __shared__ __align__(16) u16  Kt[2][BN * DHEAD];   // [key][i]
  __shared__ __align__(16) u16  Vt[2][DHEAD * BN];   // [i][key]
  __shared__ float Wl[CCH * DHEAD];
  __shared__ float Wbl[CCH];
  __shared__ __align__(16) float yl[DHEAD * 68];     // [i][row], padded

  const int b    = blockIdx.y;
  const int row0 = blockIdx.x * BM;
  const int tid  = threadIdx.x;
  const int wave = tid >> 5;
  const int lane = tid & 31;
  const int lr   = lane & 15;
  const int half = lane >> 4;
  const int bp   = ((lane ^ 16) << 2);     // ds_bpermute byte address, hoisted

  for (int i = tid; i < CCH * DHEAD; i += 128) Wl[i] = Ww[i];
  if (tid < CCH) Wbl[tid] = Wb[tid];

  // Q as B-fragment: lane = query column q = row0 + wave*16 + lr,
  // holding inter values half*8..+7 and 16+half*8..+7.
  FragAB qb;
  {
    const u32x4* qp = (const u32x4*)(Qg + ((size_t)(b * NPIX + row0 + wave*16 + lr)) * DHEAD);
    qb.u[0] = qp[half];
    qb.u[1] = qp[half + 2];
  }

  v8f acc0 = {}; v8f acc1 = {};
  float mst = -1e30f, lst = 0.f;

  const int vi = tid >> 2, vc = tid & 3;
  auto issue = [&](int mk, int buf) {
    // K chunk: 4KB contiguous, 2 x b128 per thread
    unsigned kd = (unsigned)(size_t)(&Kt[buf][0]) + (unsigned)tid * 16u;
    unsigned long long kg = (unsigned long long)(size_t)(Kg + ((size_t)(b * NPIX + mk)) * DHEAD)
                          + (unsigned long long)tid * 16ull;
    asm volatile("global_load_async_to_lds_b128 %0, %1, off" :: "v"(kd), "v"(kg) : "memory");
    asm volatile("global_load_async_to_lds_b128 %0, %1, off" :: "v"(kd + 2048u), "v"(kg + 2048ull) : "memory");
    // V chunk: 32 rows x 128B, 2 x b128 per thread
    unsigned vd = (unsigned)(size_t)(&Vt[buf][0]) + (unsigned)(vi * 128 + vc * 16);
    unsigned long long vg = (unsigned long long)(size_t)(Vg + ((size_t)(b * DHEAD + vi)) * NPIX + mk)
                          + (unsigned long long)vc * 16ull;
    asm volatile("global_load_async_to_lds_b128 %0, %1, off" :: "v"(vd), "v"(vg) : "memory");
    asm volatile("global_load_async_to_lds_b128 %0, %1, off" :: "v"(vd + 64u), "v"(vg + 64ull) : "memory");
  };

  issue(0, 0);
  const int NK = NPIX / BN;
  for (int kb = 0; kb < NK; ++kb) {
    const int cur = kb & 1;
    if (kb + 1 < NK) {
      issue((kb + 1) * BN, cur ^ 1);
      asm volatile("s_wait_asynccnt 0x4" ::: "memory");   // current chunk done, prefetch in flight
    } else {
      asm volatile("s_wait_asynccnt 0x0" ::: "memory");
    }
    __syncthreads();

    // ---- S^T = K_tile (16x32) x Q^T (32x16): 4 tiles (log2-domain logits) ----
    v8f st[4];
    #pragma unroll
    for (int t = 0; t < 4; ++t) {
      FragAB ka;
      const u32x4* kp = (const u32x4*)(&Kt[cur][(t*16 + lr) * DHEAD]);
      ka.u[0] = kp[half];
      ka.u[1] = kp[half + 2];
      v8f z = {};
      st[t] = __builtin_amdgcn_wmma_f32_16x16x32_bf16(
          false, ka.v, false, qb.v, (short)0, z, false, false);
    }

    // ---- online softmax: in-lane reduce over 32 values + one cross-half op ----
    float mx = st[0][0];
    #pragma unroll
    for (int t = 0; t < 4; ++t)
      #pragma unroll
      for (int r = 0; r < 8; ++r) mx = fmaxf(mx, st[t][r]);
    mx = fmaxf(mx, xor16_f(mx, bp));
    const float mnew = fmaxf(mst, mx);
    const float corr = exp2_raw(mst - mnew);
    mst = mnew;

    unsigned pk[4][4];
    float rs = 0.f;
    #pragma unroll
    for (int t = 0; t < 4; ++t)
      #pragma unroll
      for (int j = 0; j < 4; ++j) {
        const float e0 = exp2_raw(st[t][2*j]   - mnew);
        const float e1 = exp2_raw(st[t][2*j+1] - mnew);
        rs += e0 + e1;
        pk[t][j] = pack_bf16x2_trunc(e0, e1);
      }
    rs += xor16_f(rs, bp);
    lst = lst * corr + rs;
    #pragma unroll
    for (int r = 0; r < 8; ++r) { acc0[r] *= corr; acc1[r] *= corr; }

    // ---- y^T += V^T (16x32-keys) x P^T (32-keys x 16): 2 k-steps x 2 i-tiles ----
    #pragma unroll
    for (int kk = 0; kk < 2; ++kk) {
      FragAB pb;   // P^T B-fragment: lane = query col, keys kk*32 + half*16 .. +15
      #pragma unroll
      for (int j = 0; j < 4; ++j) {
        const unsigned lo = pk[2*kk][j];      // keys kk*32 + half*8 + 2j..2j+1
        const unsigned hi = pk[2*kk+1][j];    // keys kk*32 + 16 + half*8 + 2j..
        const unsigned send = half ? lo : hi;
        const unsigned recv = xor16_u(send, bp);
        pb.w[j]     = half ? recv : lo;
        pb.w[4 + j] = half ? hi : recv;
      }
      #pragma unroll
      for (int it = 0; it < 2; ++it) {
        FragAB va;
        const u32x4* vp = (const u32x4*)(&Vt[cur][(it*16 + lr) * BN]);
        va.u[0] = vp[kk*4 + half*2];
        va.u[1] = vp[kk*4 + half*2 + 1];
        if (it == 0)
          acc0 = __builtin_amdgcn_wmma_f32_16x16x32_bf16(
              false, va.v, false, pb.v, (short)0, acc0, false, false);
        else
          acc1 = __builtin_amdgcn_wmma_f32_16x16x32_bf16(
              false, va.v, false, pb.v, (short)0, acc1, false, false);
      }
    }
    __syncthreads();   // everyone done reading `cur` before it is refilled
  }

  // ---- normalize, stash y^T[i][row] in LDS ----
  {
    const float inv = 1.0f / lst;
    const int rl = wave*16 + lr;
    #pragma unroll
    for (int r = 0; r < 8; ++r) {
      yl[(half*8 + r) * 68 + rl]      = acc0[r] * inv;
      yl[(16 + half*8 + r) * 68 + rl] = acc1[r] * inv;
    }
  }
  __syncthreads();

  // ---- fused epilogue: out = W_w @ y + W_b + x ----
  const int nl = tid & 63;
  const int c0 = (tid >> 6) * 32;
  for (int cc = 0; cc < 32; ++cc) {
    const int c = c0 + cc;
    float a = Wbl[c];
    #pragma unroll
    for (int i = 0; i < DHEAD; ++i) a += Wl[c * DHEAD + i] * yl[i * 68 + nl];
    const size_t oidx = ((size_t)(b * CCH + c)) * NPIX + row0 + nl;
    out[oidx] = a + x[oidx];
  }
}

extern "C" void kernel_launch(void* const* d_in, const int* in_sizes, int n_in,
                              void* d_out, int out_size, void* d_ws, size_t ws_size,
                              hipStream_t stream) {
  (void)in_sizes; (void)n_in; (void)out_size; (void)ws_size;
  const float* x    = (const float*)d_in[0];
  const float* g_w  = (const float*)d_in[1];
  const float* g_b  = (const float*)d_in[2];
  const float* th_w = (const float*)d_in[3];
  const float* th_b = (const float*)d_in[4];
  const float* ph_w = (const float*)d_in[5];
  const float* ph_b = (const float*)d_in[6];
  const float* W_w  = (const float*)d_in[7];
  const float* W_b  = (const float*)d_in[8];
  float* out = (float*)d_out;

  u16* Q = (u16*)d_ws;                       // [2][9216][32] bf16 (log2e-scaled)
  u16* K = Q + (size_t)2 * NPIX * DHEAD;     // [2][9216][32] bf16
  u16* V = K + (size_t)2 * NPIX * DHEAD;     // [2][32][9216] bf16

  proj_kernel<<<dim3(NPIX / 256, 2), 256, 0, stream>>>(
      x, g_w, g_b, th_w, th_b, ph_w, ph_b, Q, K, V);
  flash_kernel<<<dim3(NPIX / BM, 2), 128, 0, stream>>>(
      Q, K, V, x, W_w, W_b, out);
}
